// Block_927712936453
// MI455X (gfx1250) — matrix operand order repaired and checked by reading, not compiled, not run on hardware
//
#include <hip/hip_runtime.h>
#include <hip/hip_bf16.h>
#include <math.h>

typedef __attribute__((ext_vector_type(16))) _Float16 v16h;
typedef __attribute__((ext_vector_type(8)))  float    v8f;
typedef unsigned int u32;
typedef __attribute__((ext_vector_type(4))) u32 u32x4;
typedef __attribute__((ext_vector_type(8))) int i32x8;
typedef __attribute__((ext_vector_type(4))) int i32x4;

#define DIM      384
#define HEADS    6
#define HEAD_DIM 64
#define HIDDEN   1536
#define NX       2048
#define NY       256
#define SEQ      2304          // NX + NY
#define BATCH    4
#define TOK      (BATCH*SEQ)   // 9216
#define ATT_SCALE 0.125f       // 64^-0.5
#define LN_EPS   1e-5f
#define KTILE    128

// token (b,s) -> flat index into d_out (new_x [4,2048,384] then new_y [4,256,384])
__device__ __forceinline__ size_t out_index(int b, int s) {
  return (s < NX) ? ((size_t)(b * NX + s) * DIM)
                  : ((size_t)BATCH * NX * DIM + (size_t)(b * NY + (s - NX)) * DIM);
}

// ---------- TDM: 2D tile (tile_k x tile_n elems, f16) global -> LDS ----------
// D# per CDNA5 ISA ch.8: group0 {count=1 | lds_addr | global_addr | type=2},
// group1 {data_size=1(2B) | tensor_dim0/1 | tile_dim0/1 | dim0_stride}.
// This toolchain (clang-23 / therock-10.0) uses the 6-arg builtin form.
__device__ __forceinline__ void tdm_load_tile(u32 lds_byte_off, const _Float16* gsrc,
                                              u32 tile_k, u32 tile_n, u32 stride_k) {
  unsigned long long ga = (unsigned long long)(uintptr_t)gsrc;
  u32x4 g0;
  g0[0] = 1u;                                             // count=1, user desc
  g0[1] = lds_byte_off;                                   // lds_addr (bytes)
  g0[2] = (u32)ga;                                        // global_addr[31:0]
  g0[3] = (u32)((ga >> 32) & 0x1FFFFFFu) | (2u << 30);    // global_addr[56:32] | type=2
  u32 td0 = stride_k;                                     // tensor_dim0 (elems)
  u32 td1 = tile_n;                                       // tensor_dim1 (elems)
  i32x8 g1;
  g1[0] = (int)(1u << 16);                                // workgroup_mask=0, data_size=1 (2B)
  g1[1] = (int)((td0 & 0xFFFFu) << 16);                   // tensor_dim0[15:0] @ bits 63:48
  g1[2] = (int)(((td0 >> 16) & 0xFFFFu) | ((td1 & 0xFFFFu) << 16));
  g1[3] = (int)(((td1 >> 16) & 0xFFFFu) | ((tile_k & 0xFFFFu) << 16)); // tile_dim0
  g1[4] = (int)(tile_n & 0xFFFFu);                        // tile_dim1 (tile_dim2=0)
  g1[5] = (int)stride_k;                                  // tensor_dim0_stride[31:0]
  g1[6] = 0;                                              // stride0[47:32], stride1[15:0]
  g1[7] = 0;
  i32x4 z4; z4[0] = 0; z4[1] = 0; z4[2] = 0; z4[3] = 0;   // groups 2/3 unused (2D tensor)
  i32x8 z8;
#pragma unroll
  for (int i = 0; i < 8; ++i) z8[i] = 0;
  __builtin_amdgcn_tensor_load_to_lds(g0, g1, z4, z4, z8, 0);  // 6-arg form
}

// ---------- WMMA fragment loaders (wave32, v_wmma_f32_16x16x32_f16) ----------
// A 16x32 (MxK): lane l: m=l&15, hi=l>>4; pair j: k=2j+16hi
__device__ __forceinline__ v16h load_frag_a(const _Float16* __restrict__ A, int lda,
                                            int row0, int k0) {
  int lane = threadIdx.x & 31;
  int m = lane & 15, hi = lane >> 4;
  const _Float16* base = A + (size_t)(row0 + m) * lda + k0;
  v16h f;
#pragma unroll
  for (int j = 0; j < 8; ++j) {
    int k = 2 * j + 16 * hi;
    f[2 * j]     = base[k];
    f[2 * j + 1] = base[k + 1];
  }
  return f;
}

// B 32x16 (KxN), B[k][n] = W[n0+n][k0+k] (transposed access of row-major W)
__device__ __forceinline__ v16h load_frag_bT(const _Float16* __restrict__ W, int ldw,
                                             int n0, int k0) {
  int lane = threadIdx.x & 31;
  int n = lane & 15, hi = lane >> 4;
  const _Float16* base = W + (size_t)(n0 + n) * ldw + k0 + 16 * hi;
  v16h f;
#pragma unroll
  for (int j = 0; j < 8; ++j) { f[2 * j] = base[2 * j]; f[2 * j + 1] = base[2 * j + 1]; }
  return f;
}

// B 32x16 (KxN), B[k][n] = M[k0+k][n0+n] (row-major M, strided)
__device__ __forceinline__ v16h load_frag_b(const _Float16* __restrict__ M, int ldm,
                                            int k0, int n0) {
  int lane = threadIdx.x & 31;
  int n = lane & 15, hi = lane >> 4;
  const _Float16* base = M + (size_t)(k0 + 16 * hi) * ldm + n0 + n;
  v16h f;
#pragma unroll
  for (int j = 0; j < 8; ++j) {
    f[2 * j]     = base[(size_t)(2 * j) * ldm];
    f[2 * j + 1] = base[(size_t)(2 * j + 1) * ldm];
  }
  return f;
}

// ---------- weight f32 -> f16 ----------
__global__ __launch_bounds__(256) void w2h_kernel(const float* __restrict__ src,
                                                  _Float16* __restrict__ dst, int n) {
  int i = blockIdx.x * 256 + threadIdx.x;
  if (i < n) dst[i] = (_Float16)src[i];
}

// ---------- LayerNorm: one wave per token, 8 tokens per block ----------
__global__ __launch_bounds__(256) void ln_kernel(
    const float* __restrict__ xp, const float* __restrict__ yp,
    const float* __restrict__ outp,               // if non-null: read d_out via out_index
    const float* __restrict__ w, const float* __restrict__ bias,
    _Float16* __restrict__ dst) {
  int wv = threadIdx.x >> 5, lane = threadIdx.x & 31;
  int t = blockIdx.x * 8 + wv;
  int b = t / SEQ, s = t % SEQ;
  const float* src = outp ? (outp + out_index(b, s))
                          : (s < NX ? xp + (size_t)(b * NX + s) * DIM
                                    : yp + (size_t)(b * NY + (s - NX)) * DIM);
  float v[12], sum = 0.f, sq = 0.f;
#pragma unroll
  for (int j = 0; j < 12; ++j) { v[j] = src[lane + 32 * j]; sum += v[j]; sq += v[j] * v[j]; }
#pragma unroll
  for (int off = 16; off >= 1; off >>= 1) {
    sum += __shfl_xor(sum, off);
    sq  += __shfl_xor(sq,  off);
  }
  float mean = sum * (1.0f / DIM);
  float var  = sq * (1.0f / DIM) - mean * mean;
  float rstd = rsqrtf(var + LN_EPS);
#pragma unroll
  for (int j = 0; j < 12; ++j) {
    int c = lane + 32 * j;
    dst[(size_t)t * DIM + c] = (_Float16)((v[j] - mean) * rstd * w[c] + bias[c]);
  }
}

// ---------- GEMM: C[M,N] = A[M,K](f16) * Wh[N,K]^T (f16), TDM-staged weights ----------
// block: 256 thr = 8 waves; block tile 256(M) x 64(N); wave tile 32 x 64.
// Double-buffered LDS weight tiles filled by the Tensor Data Mover (wave 0 issues),
// synchronized with s_wait_tensorcnt + block barrier; TDM overlaps WMMA compute.
// EPI 0: scatter qkv -> q/k/v [B,H,SEQ,64] f16
// EPI 1: d_out = residual(x|y) + acc + bias
// EPI 2: outH  = gelu_erf(acc + bias)  (f16)
// EPI 3: d_out += acc + bias
template <int EPI>
__global__ __launch_bounds__(256) void gemm_kernel(
    const _Float16* __restrict__ A, const _Float16* __restrict__ Wh,
    const float* __restrict__ bias, int M, int N, int K,
    float* outF, _Float16* outH,
    const float* __restrict__ resx, const float* __restrict__ resy,
    _Float16* qh, _Float16* kh, _Float16* vh) {
  __shared__ _Float16 lds_w[2][64 * KTILE];   // 2 x 16 KB
  int wv = threadIdx.x >> 5, lane = threadIdx.x & 31;
  int row0 = blockIdx.x * 256 + wv * 32;
  int n0   = blockIdx.y * 64;
  v8f acc[2][4];
#pragma unroll
  for (int u = 0; u < 2; ++u)
#pragma unroll
    for (int i = 0; i < 4; ++i) acc[u][i] = (v8f)(0.0f);

  const int nstages = K / KTILE;
  const _Float16* wsrc = Wh + (size_t)n0 * K;
  if (wv == 0) {
    tdm_load_tile((u32)(uintptr_t)&lds_w[0][0], wsrc, KTILE, 64, (u32)K);
    __builtin_amdgcn_s_wait_tensorcnt(0);
  }
  __syncthreads();

  for (int st = 0; st < nstages; ++st) {
    int k0 = st * KTILE;
    int cur = st & 1;
    if (st + 1 < nstages) {
      if (wv == 0)   // async: TDM fills the other buffer while we compute
        tdm_load_tile((u32)(uintptr_t)&lds_w[cur ^ 1][0], wsrc + k0 + KTILE,
                      KTILE, 64, (u32)K);
      __builtin_prefetch(A + (size_t)(row0 + lane) * K + k0 + KTILE, 0, 1);
    }
#pragma unroll
    for (int kk = 0; kk < KTILE; kk += 32) {
      v16h a0 = load_frag_a(A, K, row0,      k0 + kk);
      v16h a1 = load_frag_a(A, K, row0 + 16, k0 + kk);
#pragma unroll
      for (int i = 0; i < 4; ++i) {
        const _Float16* base = &lds_w[cur][0] +
            (16 * i + (lane & 15)) * KTILE + kk + 16 * (lane >> 4);
        v16h bfr;
#pragma unroll
        for (int j = 0; j < 8; ++j) { bfr[2 * j] = base[2 * j]; bfr[2 * j + 1] = base[2 * j + 1]; }
        acc[0][i] = __builtin_amdgcn_wmma_f32_16x16x32_f16(
            false, a0, false, bfr, (short)0, acc[0][i], false, false);
        acc[1][i] = __builtin_amdgcn_wmma_f32_16x16x32_f16(
            false, a1, false, bfr, (short)0, acc[1][i], false, false);
      }
    }
    if (wv == 0 && st + 1 < nstages) __builtin_amdgcn_s_wait_tensorcnt(0);
    __syncthreads();
  }

  // epilogue: D tile layout -> lane: n = lane&15, row m = r + 8*(lane>>4)
  int hi = lane >> 4, nl = lane & 15;
#pragma unroll
  for (int u = 0; u < 2; ++u) {
#pragma unroll
    for (int i = 0; i < 4; ++i) {
      int n = n0 + 16 * i + nl;
#pragma unroll
      for (int r = 0; r < 8; ++r) {
        int m = row0 + 16 * u + r + 8 * hi;
        float val = acc[u][i][r];
        if (EPI == 0) {
          int t = n / DIM, rem = n % DIM;
          int hh = rem >> 6, d = rem & 63;
          int b = m / SEQ, s = m % SEQ;
          _Float16* dst = (t == 0) ? qh : (t == 1) ? kh : vh;
          dst[(((size_t)b * HEADS + hh) * SEQ + s) * HEAD_DIM + d] = (_Float16)val;
        } else if (EPI == 1) {
          int b = m / SEQ, s = m % SEQ;
          float res = (s < NX) ? resx[(size_t)(b * NX + s) * DIM + n]
                               : resy[(size_t)(b * NY + (s - NX)) * DIM + n];
          outF[out_index(b, s) + n] = res + val + bias[n];
        } else if (EPI == 2) {
          float hv = val + bias[n];
          hv = 0.5f * hv * (1.0f + erff(hv * 0.70710678118654752f));  // exact-erf GELU
          outH[(size_t)m * HIDDEN + n] = (_Float16)hv;
        } else {  // EPI == 3
          int b = m / SEQ, s = m % SEQ;
          size_t idx = out_index(b, s) + n;
          outF[idx] = outF[idx] + val + bias[n];
        }
      }
    }
  }
}

// ---------- Flash attention: wave = 16 q-rows x full head (D=64), online softmax ----------
__global__ __launch_bounds__(256) void attn_kernel(
    const _Float16* __restrict__ qh, const _Float16* __restrict__ kh,
    const _Float16* __restrict__ vh, _Float16* __restrict__ att,
    int q_off, int k_len) {
  __shared__ _Float16 lds_p[8][16][32];  // per-wave P transpose buffer (8 KB)
  int wv = threadIdx.x >> 5, lane = threadIdx.x & 31;
  int hi = lane >> 4, nl = lane & 15;
  int bh = blockIdx.y;
  int b = bh / HEADS, h = bh % HEADS;
  const _Float16* Qp = qh + (size_t)bh * SEQ * HEAD_DIM;
  const _Float16* Kp = kh + (size_t)bh * SEQ * HEAD_DIM;
  const _Float16* Vp = vh + (size_t)bh * SEQ * HEAD_DIM;
  int qs = q_off + blockIdx.x * 128 + wv * 16;

  v16h aq0 = load_frag_a(Qp, HEAD_DIM, qs, 0);    // Q d=0..31
  v16h aq1 = load_frag_a(Qp, HEAD_DIM, qs, 32);   // Q d=32..63
  v8f o[4];
#pragma unroll
  for (int i = 0; i < 4; ++i) o[i] = (v8f)(0.0f);
  float rmax[8], rsum[8];
#pragma unroll
  for (int r = 0; r < 8; ++r) { rmax[r] = -1e30f; rsum[r] = 0.0f; }

  for (int k0 = 0; k0 < k_len; k0 += 32) {
    // S = (Q K^T) * scale  : two 16x16 logit tiles
    v8f z = (v8f)(0.0f), s0, s1;
    v16h bk;
    bk = load_frag_bT(Kp, HEAD_DIM, k0, 0);
    s0 = __builtin_amdgcn_wmma_f32_16x16x32_f16(false, aq0, false, bk, (short)0, z, false, false);
    bk = load_frag_bT(Kp, HEAD_DIM, k0, 32);
    s0 = __builtin_amdgcn_wmma_f32_16x16x32_f16(false, aq1, false, bk, (short)0, s0, false, false);
    bk = load_frag_bT(Kp, HEAD_DIM, k0 + 16, 0);
    s1 = __builtin_amdgcn_wmma_f32_16x16x32_f16(false, aq0, false, bk, (short)0, z, false, false);
    bk = load_frag_bT(Kp, HEAD_DIM, k0 + 16, 32);
    s1 = __builtin_amdgcn_wmma_f32_16x16x32_f16(false, aq1, false, bk, (short)0, s1, false, false);

    // online softmax; row m=r+8hi lives in a 16-lane group -> xor offs 8,4,2,1
#pragma unroll
    for (int r = 0; r < 8; ++r) {
      float l0 = s0[r] * ATT_SCALE, l1 = s1[r] * ATT_SCALE;
      float tm = fmaxf(l0, l1);
#pragma unroll
      for (int off = 8; off >= 1; off >>= 1) tm = fmaxf(tm, __shfl_xor(tm, off));
      float nm = fmaxf(rmax[r], tm);
      float cf = expf(rmax[r] - nm);
      float p0 = expf(l0 - nm), p1 = expf(l1 - nm);
      float ps = p0 + p1;
#pragma unroll
      for (int off = 8; off >= 1; off >>= 1) ps += __shfl_xor(ps, off);
      rsum[r] = rsum[r] * cf + ps;
      rmax[r] = nm;
      s0[r] = p0; s1[r] = p1;
#pragma unroll
      for (int i = 0; i < 4; ++i) o[i][r] *= cf;
    }

    // transpose P (C layout -> A layout) through wave-private LDS
#pragma unroll
    for (int r = 0; r < 8; ++r) {
      lds_p[wv][r + 8 * hi][nl]      = (_Float16)s0[r];
      lds_p[wv][r + 8 * hi][16 + nl] = (_Float16)s1[r];
    }
    asm volatile("s_wait_dscnt 0" ::: "memory");  // same-wave DS store->load ordering
    v16h pa;
    {
      int m = lane & 15, h2 = lane >> 4;
#pragma unroll
      for (int j = 0; j < 8; ++j) {
        int k = 2 * j + 16 * h2;
        pa[2 * j]     = lds_p[wv][m][k];
        pa[2 * j + 1] = lds_p[wv][m][k + 1];
      }
    }
    // O += P V
#pragma unroll
    for (int i = 0; i < 4; ++i) {
      v16h bv = load_frag_b(Vp, HEAD_DIM, k0, 16 * i);
      o[i] = __builtin_amdgcn_wmma_f32_16x16x32_f16(false, pa, false, bv, (short)0, o[i], false, false);
    }
  }

#pragma unroll
  for (int r = 0; r < 8; ++r) {
    float inv = 1.0f / rsum[r];
#pragma unroll
    for (int i = 0; i < 4; ++i) o[i][r] *= inv;
  }
  // store token-major f16: att[(b*SEQ+s)*DIM + h*64 + d]
#pragma unroll
  for (int i = 0; i < 4; ++i) {
    int d = 16 * i + nl;
#pragma unroll
    for (int r = 0; r < 8; ++r) {
      int s = qs + r + 8 * hi;
      att[((size_t)b * SEQ + s) * DIM + h * HEAD_DIM + d] = (_Float16)o[i][r];
    }
  }
}

extern "C" void kernel_launch(void* const* d_in, const int* in_sizes, int n_in,
                              void* d_out, int out_size, void* d_ws, size_t ws_size,
                              hipStream_t stream) {
  const float* x     = (const float*)d_in[0];
  const float* y     = (const float*)d_in[1];
  const float* n1w   = (const float*)d_in[2];
  const float* n1b   = (const float*)d_in[3];
  const float* n2w   = (const float*)d_in[4];
  const float* n2b   = (const float*)d_in[5];
  const float* qkvw  = (const float*)d_in[6];
  const float* projw = (const float*)d_in[7];
  const float* projb = (const float*)d_in[8];
  const float* fc1w  = (const float*)d_in[9];
  const float* fc1b  = (const float*)d_in[10];
  const float* fc2w  = (const float*)d_in[11];
  const float* fc2b  = (const float*)d_in[12];
  float* out = (float*)d_out;

  // scratch layout (f16), ~68 MB total
  _Float16* ws = (_Float16*)d_ws;
  size_t off = 0;
  _Float16* cat_h = ws + off; off += (size_t)TOK * DIM;      // LN1 out (reused for LN2)
  _Float16* qh    = ws + off; off += (size_t)BATCH * HEADS * SEQ * HEAD_DIM;
  _Float16* kh    = ws + off; off += (size_t)BATCH * HEADS * SEQ * HEAD_DIM;
  _Float16* vh    = ws + off; off += (size_t)BATCH * HEADS * SEQ * HEAD_DIM;
  _Float16* att_h = ws + off; off += (size_t)TOK * DIM;
  _Float16* h_h   = ws + off; off += (size_t)TOK * HIDDEN;
  _Float16* whq   = ws + off; off += (size_t)3 * DIM * DIM;       // qkv_w f16
  _Float16* whp   = ws + off; off += (size_t)DIM * DIM;           // proj_w f16
  _Float16* wh1   = ws + off; off += (size_t)HIDDEN * DIM;        // fc1_w f16
  _Float16* wh2   = ws + off; off += (size_t)DIM * HIDDEN;        // fc2_w f16
  (void)ws_size; (void)in_sizes; (void)n_in; (void)out_size;

  // 0) weights -> f16 (TDM moves raw f16 tiles; no conversion in DMA path)
  w2h_kernel<<<(3 * DIM * DIM + 255) / 256, 256, 0, stream>>>(qkvw, whq, 3 * DIM * DIM);
  w2h_kernel<<<(DIM * DIM + 255) / 256, 256, 0, stream>>>(projw, whp, DIM * DIM);
  w2h_kernel<<<(HIDDEN * DIM + 255) / 256, 256, 0, stream>>>(fc1w, wh1, HIDDEN * DIM);
  w2h_kernel<<<(DIM * HIDDEN + 255) / 256, 256, 0, stream>>>(fc2w, wh2, DIM * HIDDEN);

  // 1) LN1 over cat(x, y)
  ln_kernel<<<TOK / 8, 256, 0, stream>>>(x, y, nullptr, n1w, n1b, cat_h);
  // 2) QKV projection, scatter to [B,H,S,64]
  gemm_kernel<0><<<dim3(TOK / 256, (3 * DIM) / 64), 256, 0, stream>>>(
      cat_h, whq, nullptr, TOK, 3 * DIM, DIM,
      nullptr, nullptr, nullptr, nullptr, qh, kh, vh);
  // 3) self-attn: x queries over first NX keys
  attn_kernel<<<dim3(NX / 128, BATCH * HEADS), 256, 0, stream>>>(qh, kh, vh, att_h, 0, NX);
  // 4) cross-attn: y queries over all SEQ keys
  attn_kernel<<<dim3(NY / 128, BATCH * HEADS), 256, 0, stream>>>(qh, kh, vh, att_h, NX, SEQ);
  // 5) output projection + residual -> d_out
  gemm_kernel<1><<<dim3(TOK / 256, DIM / 64), 256, 0, stream>>>(
      att_h, whp, projb, TOK, DIM, DIM,
      out, nullptr, x, y, nullptr, nullptr, nullptr);
  // 6) LN2 on d_out
  ln_kernel<<<TOK / 8, 256, 0, stream>>>(nullptr, nullptr, out, n2w, n2b, cat_h);
  // 7) fc1 + exact GELU
  gemm_kernel<2><<<dim3(TOK / 256, HIDDEN / 64), 256, 0, stream>>>(
      cat_h, wh1, fc1b, TOK, HIDDEN, DIM,
      nullptr, h_h, nullptr, nullptr, nullptr, nullptr, nullptr);
  // 8) fc2 + residual accumulate into d_out
  gemm_kernel<3><<<dim3(TOK / 256, DIM / 64), 256, 0, stream>>>(
      h_h, wh2, fc2b, TOK, DIM, HIDDEN,
      out, nullptr, nullptr, nullptr, nullptr, nullptr, nullptr);
}
